// CNN3D_32770600468665
// MI455X (gfx1250) — matrix operand order repaired and checked by reading, not compiled, not run on hardware
//
#include <hip/hip_runtime.h>
#include <hip/hip_bf16.h>
#include <math.h>

typedef __attribute__((ext_vector_type(16))) __bf16 v16bf;
typedef __attribute__((ext_vector_type(8)))  float  v8f;

#define RR     40
#define NS     64000        // 40^3 spatial sites
#define FDIM   54
#define FPAD   64
#define FSTR   72           // padded LDS row stride (halves): 144 B -> conflict-free B reads
#define NB     8
#define NP1    400
#define NP2    64
#define COUT   128
#define NGRP   2000         // NS / 32 site-groups per batch

// ---- workspace layout (bytes) ----
#define SUB_OFF   ((size_t)0)                       // 8*3 floats
#define WPK_OFF   ((size_t)512)                     // 27*2*128*2*16 ushort = 442368 B
#define LAT_OFF   ((size_t)(448*1024))              // 458752 (>= 512+442368)
#define LAT_BYTES ((size_t)NB*NS*FPAD*2)            // 65,536,000
#define PART_OFF  (LAT_OFF + LAT_BYTES)             // 8*2000 floats

__device__ __forceinline__ unsigned short f2bf(float f) {
    unsigned int u = __float_as_uint(f);
    u += 0x7fffu + ((u >> 16) & 1u);                // round-to-nearest-even
    return (unsigned short)(u >> 16);
}

// ---------------- kernel 1: per-batch bbox center (sub) ----------------
__global__ void k_sub(const float* __restrict__ pos1, const float* __restrict__ valid1,
                      float* __restrict__ sub) {
    int b = blockIdx.x, t = threadIdx.x;
    __shared__ float smx[3][128], smn[3][128];
    float mx[3] = {-1e30f, -1e30f, -1e30f}, mn[3] = {1e30f, 1e30f, 1e30f};
    for (int i = t; i < NP1; i += 128) {
        const float* p = pos1 + ((size_t)b * NP1 + i) * 3;
        float v = valid1[b * NP1 + i];
        float c[3] = {p[0] * v, p[1] * v, p[2] * v};
        float s = c[0] + c[1] + c[2];
        float amx = (s == 0.f) ? -1e10f : 0.f;
        float amn = (s == 0.f) ?  1e10f : 0.f;
        #pragma unroll
        for (int k = 0; k < 3; ++k) {
            mx[k] = fmaxf(mx[k], c[k] + amx);
            mn[k] = fminf(mn[k], c[k] + amn);
        }
    }
    #pragma unroll
    for (int k = 0; k < 3; ++k) { smx[k][t] = mx[k]; smn[k][t] = mn[k]; }
    __syncthreads();
    for (int st = 64; st > 0; st >>= 1) {
        if (t < st) {
            #pragma unroll
            for (int k = 0; k < 3; ++k) {
                smx[k][t] = fmaxf(smx[k][t], smx[k][t + st]);
                smn[k][t] = fminf(smn[k][t], smn[k][t + st]);
            }
        }
        __syncthreads();
    }
    if (t < 3) sub[b * 3 + t] = smn[t][0] + (smx[t][0] - smn[t][0]) * 0.5f;
}

// ---------------- kernel 2: zero the bf16 lattice ----------------
__global__ void k_zero(uint4* __restrict__ lat) {
    size_t i = (size_t)blockIdx.x * blockDim.x + threadIdx.x;   // LAT_BYTES/16 threads
    lat[i] = make_uint4(0u, 0u, 0u, 0u);
}

// ---------------- kernel 3: repack conv_w into WMMA A-fragment order ----------------
// layout: (((tap*2 + kc)*128 + c)*2 + khalf)*16 + j
// lane's 16 halves j: K_local = khalf*16 + j
__global__ void k_pack(const float* __restrict__ cw, unsigned short* __restrict__ wpk) {
    int i = blockIdx.x * blockDim.x + threadIdx.x;
    if (i >= 27 * 2 * COUT * 2 * 16) return;
    int j     = i & 15;
    int khalf = (i >> 4) & 1;
    int c     = (i >> 5) & 127;
    int kc    = (i >> 12) & 1;
    int tap   = i >> 13;
    int kl    = khalf * 16 + j;
    int f     = kc * 32 + kl;
    float v   = (f < FDIM) ? cw[((size_t)c * FDIM + f) * 27 + tap] : 0.f;
    wpk[i]    = f2bf(v);
}

// ---------------- kernel 4: scatter points into post-rotation channels-last lattice ----------------
// (ix,iy,iz) -> (d,h,w) = (ix, 39-iy, 39-iz)
__global__ void k_scatter(const float* __restrict__ pos, const float* __restrict__ h,
                          const float* __restrict__ valid, int npts,
                          const float* __restrict__ sub, unsigned short* __restrict__ X) {
    int i = blockIdx.x * blockDim.x + threadIdx.x;
    if (i >= NB * npts) return;
    int b = i / npts, p = i - b * npts;
    float v = valid[b * npts + p];
    const float* pp = pos + ((size_t)b * npts + p) * 3;
    const float* sb = sub + b * 3;
    int ix = (int)floorf((pp[0] * v - sb[0] + 10.f) * 2.f);
    int iy = (int)floorf((pp[1] * v - sb[1] + 10.f) * 2.f);
    int iz = (int)floorf((pp[2] * v - sb[2] + 10.f) * 2.f);
    if (ix < 0 || ix > RR - 1 || iy < 0 || iy > RR - 1 || iz < 0 || iz > RR - 1) return;
    int d = ix, hh = (RR - 1) - iy, ww = (RR - 1) - iz;
    size_t base = (((size_t)b * NS) + (size_t)d * 1600 + hh * 40 + ww) * FPAD;
    const float* ft = h + ((size_t)b * npts + p) * FDIM;
    #pragma unroll
    for (int f = 0; f < FDIM; f += 2) {
        unsigned int lo = f2bf(ft[f] * v), hi = f2bf(ft[f + 1] * v);
        *(unsigned int*)(X + base + f) = lo | (hi << 16);
    }
}

// ---------------- kernel 5: fused implicit-GEMM conv + bias + ReLU + fc dot ----------------
// block = 256 thr (8 waves). wave w -> channels [16w,16w+16); block -> batch blockIdx.y,
// 32 spatial sites s0..s0+31 (two 16-wide N tiles sharing each A fragment).
// Double-buffered LDS (padded rows, conflict-free B reads); one barrier per tap;
// next tap's global load prefetched into regs with compile-time immediate offsets.
__global__ __launch_bounds__(256) void
k_conv_gemm(const unsigned short* __restrict__ X, const unsigned short* __restrict__ wpk,
            const float* __restrict__ conv_b, const float* __restrict__ fc_w,
            float* __restrict__ partial) {
    __shared__ __align__(64) unsigned short ldsB[2][32 * FSTR];   // 2 x 4.5KB padded tiles
    __shared__ float wsum[8];

    int tid = threadIdx.x, lane = tid & 31, wave = tid >> 5;
    int b = blockIdx.y, g = blockIdx.x;
    int s0 = g * 32;

    // cooperative-load mapping: site ln (0..31), features f8..f8+7 (one b128 per thread)
    int ln = tid >> 3;
    int f8 = (tid & 7) << 3;
    int s = s0 + ln;
    int d = s / 1600;
    int r = s - d * 1600;
    int hh = r / 40;
    int ww = r - hh * 40;

    // 27-bit tap validity mask (tap = kd*9 + kh*3 + kw)
    unsigned vmask = 0;
    #pragma unroll
    for (int kd = 0; kd < 3; ++kd)
        #pragma unroll
        for (int kh = 0; kh < 3; ++kh)
            #pragma unroll
            for (int kw = 0; kw < 3; ++kw) {
                int dd = d + kd - 1, dh = hh + kh - 1, dw = ww + kw - 1;
                if (dd >= 0 && dd < RR && dh >= 0 && dh < RR && dw >= 0 && dw < RR)
                    vmask |= 1u << (kd * 9 + kh * 3 + kw);
            }

    // base pointer at (d-1, h-1, w-1); every tap is a compile-time immediate offset
    const unsigned short* gp =
        X + ((((size_t)b * NS) + (size_t)(d - 1) * 1600 + (hh - 1) * 40 + (ww - 1)) * FPAD) + f8;

    int col  = lane & 15;      // N column / A row
    int krow = lane >> 4;      // K half selector

    v8f acc0 = {0.f, 0.f, 0.f, 0.f, 0.f, 0.f, 0.f, 0.f};
    v8f acc1 = {0.f, 0.f, 0.f, 0.f, 0.f, 0.f, 0.f, 0.f};

    // prime buffer 0 with tap 0
    {
        uint4 vv = make_uint4(0u, 0u, 0u, 0u);
        if (vmask & 1u) vv = *(const uint4*)(gp);
        *(uint4*)(&ldsB[0][ln * FSTR + f8]) = vv;
    }

    #pragma unroll
    for (int tap = 0; tap < 27; ++tap) {
        __syncthreads();
        // prefetch next tap into registers while we crunch this one
        uint4 nxt = make_uint4(0u, 0u, 0u, 0u);
        if (tap < 26) {
            const int kd = (tap + 1) / 9, kh = ((tap + 1) % 9) / 3, kw = (tap + 1) % 3;
            const int toff = (kd * 1600 + kh * 40 + kw) * FPAD;   // compile-time
            if ((vmask >> (tap + 1)) & 1u) nxt = *(const uint4*)(gp + toff);
        }

        const unsigned short* buf = ldsB[tap & 1];
        #pragma unroll
        for (int kc = 0; kc < 2; ++kc) {
            const v16bf a = *reinterpret_cast<const v16bf*>(
                wpk + ((((size_t)(tap * 2 + kc) * COUT + wave * 16 + col) * 2 + krow) << 4));
            const v16bf b0 = *reinterpret_cast<const v16bf*>(
                buf + col * FSTR + kc * 32 + krow * 16);
            const v16bf b1 = *reinterpret_cast<const v16bf*>(
                buf + (16 + col) * FSTR + kc * 32 + krow * 16);
            acc0 = __builtin_amdgcn_wmma_f32_16x16x32_bf16(
                false, a, false, b0, (short)0, acc0, false, false);
            acc1 = __builtin_amdgcn_wmma_f32_16x16x32_bf16(
                false, a, false, b1, (short)0, acc1, false, false);
        }

        if (tap < 26)
            *(uint4*)(&ldsB[(tap + 1) & 1][ln * FSTR + f8]) = nxt;
    }

    // epilogue: bias + relu + fc-weight dot for both N tiles
    float sum = 0.f;
    int sa = s0 + col, sb2 = s0 + 16 + col;
    #pragma unroll
    for (int i = 0; i < 8; ++i) {
        int c = wave * 16 + krow * 8 + i;
        float cb = conv_b[c];
        float v0 = acc0[i] + cb; v0 = v0 > 0.f ? v0 : 0.f;
        float v1 = acc1[i] + cb; v1 = v1 > 0.f ? v1 : 0.f;
        sum += v0 * fc_w[(size_t)c * NS + sa] + v1 * fc_w[(size_t)c * NS + sb2];
    }
    for (int off = 16; off > 0; off >>= 1) sum += __shfl_down(sum, off, 32);
    if (lane == 0) wsum[wave] = sum;
    __syncthreads();
    if (tid == 0) {
        float t = 0.f;
        #pragma unroll
        for (int i = 0; i < 8; ++i) t += wsum[i];
        partial[(size_t)b * NGRP + g] = t;
    }
}

// ---------------- kernel 6: deterministic final reduction + sigmoid ----------------
__global__ void k_final(const float* __restrict__ partial, const float* __restrict__ fc_b,
                        float* __restrict__ out) {
    int b = blockIdx.x, t = threadIdx.x;
    __shared__ float sred[256];
    float a = 0.f;
    for (int i = t; i < NGRP; i += 256) a += partial[(size_t)b * NGRP + i];
    sred[t] = a;
    __syncthreads();
    for (int st = 128; st > 0; st >>= 1) {
        if (t < st) sred[t] += sred[t + st];
        __syncthreads();
    }
    if (t == 0) {
        float z = sred[0] + fc_b[0];
        out[b] = 1.f / (1.f + expf(-z));
    }
}

extern "C" void kernel_launch(void* const* d_in, const int* in_sizes, int n_in,
                              void* d_out, int out_size, void* d_ws, size_t ws_size,
                              hipStream_t stream) {
    const float* pos1   = (const float*)d_in[0];
    const float* pos2   = (const float*)d_in[1];
    const float* h1     = (const float*)d_in[2];
    const float* h2     = (const float*)d_in[3];
    const float* valid1 = (const float*)d_in[4];
    const float* valid2 = (const float*)d_in[5];
    const float* conv_w = (const float*)d_in[6];
    const float* conv_b = (const float*)d_in[7];
    const float* fc_w   = (const float*)d_in[8];
    const float* fc_b   = (const float*)d_in[9];
    float* out = (float*)d_out;

    char* ws = (char*)d_ws;
    float*          sub  = (float*)(ws + SUB_OFF);
    unsigned short* wpk  = (unsigned short*)(ws + WPK_OFF);
    unsigned short* lat  = (unsigned short*)(ws + LAT_OFF);
    float*          part = (float*)(ws + PART_OFF);

    k_sub<<<NB, 128, 0, stream>>>(pos1, valid1, sub);
    k_zero<<<(unsigned)(LAT_BYTES / 16 / 256), 256, 0, stream>>>((uint4*)lat);
    k_pack<<<(27 * 2 * COUT * 2 * 16 + 255) / 256, 256, 0, stream>>>(conv_w, wpk);
    k_scatter<<<(NB * NP1 + 255) / 256, 256, 0, stream>>>(pos1, h1, valid1, NP1, sub, lat);
    k_scatter<<<(NB * NP2 + 255) / 256, 256, 0, stream>>>(pos2, h2, valid2, NP2, sub, lat);
    k_conv_gemm<<<dim3(NGRP, NB), 256, 0, stream>>>(lat, wpk, conv_b, fc_w, part);
    k_final<<<NB, 256, 0, stream>>>(part, fc_b, out);
}
